// Lstm_10075993276706
// MI455X (gfx1250) — matrix operand order repaired and checked by reading, not compiled, not run on hardware
//
#include <hip/hip_runtime.h>

#define T_STEPS 200
#define BATCH   256
#define VOCAB   256
#define EMBD    256
#define NNODE   1024
#define GATES   4096   // 4*NNODE
#define OUTN    1024

typedef __bf16 bf16_t;
typedef __attribute__((ext_vector_type(8)))  __bf16 v8bf;
typedef __attribute__((ext_vector_type(16))) __bf16 v16bf;
typedef __attribute__((ext_vector_type(8)))  float  v8f;

__device__ __forceinline__ v8f wmma_bf16(v16bf a, v16bf b, v8f c) {
  // D = A(16x32 bf16) * B(32x16 bf16) + C(16x16 f32)
  return __builtin_amdgcn_wmma_f32_16x16x32_bf16(false, a, false, b, (short)0, c,
                                                 false, false);
}

__device__ __forceinline__ v16bf frag_cat(v8bf lo, v8bf hi) {
  return __builtin_shufflevector(lo, hi, 0, 1, 2, 3, 4, 5, 6, 7, 8, 9, 10, 11,
                                 12, 13, 14, 15);
}

// Branchless fast transcendentals on HW v_exp_f32 / v_rcp_f32.
// sigmoid(+-inf) and tanh(+-inf) are exact: exp2->{0,inf}, rcp(inf)=0.
#define LOG2E 1.44269504088896340736f
__device__ __forceinline__ float fast_sigmoid(float x) {
  float e = __builtin_amdgcn_exp2f(x * -LOG2E);
  return __builtin_amdgcn_rcpf(1.0f + e);
}
__device__ __forceinline__ float fast_tanh(float x) {
  float e = __builtin_amdgcn_exp2f(x * (-2.0f * LOG2E));
  return 2.0f * __builtin_amdgcn_rcpf(1.0f + e) - 1.0f;
}

// ---------------------------------------------------------------------------
// flat f32 -> bf16 convert
// ---------------------------------------------------------------------------
__global__ void k_cvt_bf16(const float* __restrict__ in, bf16_t* __restrict__ out,
                           int n) {
  int i = blockIdx.x * blockDim.x + threadIdx.x;
  if (i < n) out[i] = (bf16_t)in[i];
}

// in: (K,N) f32 row-major  ->  out: (N,K) bf16 row-major (transposed weights)
__global__ void k_transpose_cvt(const float* __restrict__ in,
                                bf16_t* __restrict__ out, int K, int N) {
  long i = (long)blockIdx.x * blockDim.x + threadIdx.x;
  long total = (long)K * N;
  if (i < total) {
    int n = (int)(i / K);
    int k = (int)(i % K);
    out[i] = (bf16_t)in[(long)k * N + n];
  }
}

// ---------------------------------------------------------------------------
// Generic bf16 WMMA GEMM:  C = act(A(M,K) @ Bt(N,K)^T + bias)
// 256 thr = 8 waves; tile 128(M) x 64(N); K chunks of 32; double-buffered LDS.
// Each wave: 16 rows x 4 col-tiles -> 4 accumulators, A-fragment reuse x4.
// ---------------------------------------------------------------------------
template <int RELU, int OUT_BF16>
__global__ __launch_bounds__(256) void k_gemm_wmma(
    const bf16_t* __restrict__ A, const bf16_t* __restrict__ Bt,
    const float* __restrict__ bias, void* __restrict__ Cout, int M, int N,
    int K) {
  __shared__ __align__(16) bf16_t sA[2][128 * 32];
  __shared__ __align__(16) bf16_t sB[2][64 * 32];
  const int tid = threadIdx.x;
  const int lane = tid & 31, wave = tid >> 5;
  const int row0 = blockIdx.x * 128;
  const int col0 = blockIdx.y * 64;

  v8f acc[4] = {};

  const int ar = tid >> 1, ap_ = (tid & 1) * 16;  // A staging: row, 16-elem half
  const int br = tid >> 2, bp_ = (tid & 3) * 8;   // B staging: row, 8-elem part

  auto stageA = [&](int kc, int buf) {
    const v8bf* s = (const v8bf*)(A + (size_t)(row0 + ar) * K + kc + ap_);
    v8bf* d = (v8bf*)(sA[buf] + ar * 32 + ap_);
    d[0] = s[0];
    d[1] = s[1];
  };
  auto stageB = [&](int kc, int buf) {
    *(v8bf*)(sB[buf] + br * 32 + bp_) =
        *(const v8bf*)(Bt + (size_t)(col0 + br) * K + kc + bp_);
  };

  stageA(0, 0);
  stageB(0, 0);
  __syncthreads();
  for (int kc = 0; kc < K; kc += 32) {
    const int cur = (kc >> 5) & 1, nxt = cur ^ 1;
    if (kc + 32 < K) {  // prefetch next chunk into the other buffer
      stageA(kc + 32, nxt);
      stageB(kc + 32, nxt);
    }
    // A fragment: contiguous K order (K 0..15 lanes 0-15, K 16..31 lanes 16-31)
    const bf16_t* ap2 =
        sA[cur] + (wave * 16 + (lane & 15)) * 32 + ((lane >> 4) * 16);
    v16bf afrag = frag_cat(*(const v8bf*)ap2, *(const v8bf*)(ap2 + 8));
    // Four B fragments, then four back-to-back WMMAs.
    v16bf bf[4];
#pragma unroll
    for (int j = 0; j < 4; ++j)
      bf[j] = *(const v16bf*)(sB[cur] + (j * 16 + (lane & 15)) * 32 +
                              ((lane >> 4) * 16));
#pragma unroll
    for (int j = 0; j < 4; ++j) acc[j] = wmma_bf16(afrag, bf[j], acc[j]);
    __syncthreads();
  }

  const int jl = lane & 15;
  const int rb = row0 + wave * 16 + ((lane >> 4) * 8);
  float bb[4];
#pragma unroll
  for (int j = 0; j < 4; ++j)
    bb[j] = bias ? bias[col0 + j * 16 + jl] : 0.0f;
#pragma unroll
  for (int v = 0; v < 8; ++v) {
    size_t rowoff = (size_t)(rb + v) * N + col0 + jl;
#pragma unroll
    for (int j = 0; j < 4; ++j) {
      float x = acc[j][v] + bb[j];
      if (RELU) x = fmaxf(x, 0.0f);
      if (OUT_BF16)
        ((bf16_t*)Cout)[rowoff + j * 16] = (bf16_t)x;
      else
        ((float*)Cout)[rowoff + j * 16] = x;
    }
  }
}

// ---------------------------------------------------------------------------
// Fused LSTM step: z = [x, h_prev] @ W^T + b  (4 gates), gate math, c/h update.
// Workgroup tile: 128 batch rows x 16 hidden cols x 4 gates. Grid (2, 64).
// Double-buffered LDS; branchless HW-transcendental gate math.
// ---------------------------------------------------------------------------
__global__ __launch_bounds__(256) void k_lstm_step(
    const bf16_t* __restrict__ x,      // (BATCH, Din) bf16
    const bf16_t* __restrict__ hprev,  // (BATCH, NNODE) bf16
    const bf16_t* __restrict__ Wt,     // (4*NNODE, Din+NNODE) bf16
    const float* __restrict__ bias,    // (4*NNODE) f32
    float* __restrict__ c,             // (BATCH, NNODE) f32, updated in place
    bf16_t* __restrict__ hout,         // (BATCH, NNODE) bf16
    int Din) {
  const int K = Din + NNODE;
  __shared__ __align__(16) bf16_t sA[2][128 * 32];
  __shared__ __align__(16) bf16_t sB[2][64 * 32];  // 4 gates x 16 cols x 32 K
  const int tid = threadIdx.x;
  const int lane = tid & 31, wave = tid >> 5;
  const int row0 = blockIdx.x * 128;  // batch rows
  const int col0 = blockIdx.y * 16;   // hidden column block

  v8f acc[4] = {};

  const int ar = tid >> 1, ap_ = (tid & 1) * 16;
  const int br = tid >> 2, bp_ = (tid & 3) * 8;
  const int bg = br >> 4;  // gate index of this B staging row
  const int bc = br & 15;  // column within block

  auto stageA = [&](int kc, int buf) {
    const bf16_t* src =
        (kc < Din) ? (x + (size_t)(row0 + ar) * Din + kc + ap_)
                   : (hprev + (size_t)(row0 + ar) * NNODE + (kc - Din) + ap_);
    const v8bf* s = (const v8bf*)src;
    v8bf* d = (v8bf*)(sA[buf] + ar * 32 + ap_);
    d[0] = s[0];
    d[1] = s[1];
  };
  auto stageB = [&](int kc, int buf) {
    *(v8bf*)(sB[buf] + br * 32 + bp_) =
        *(const v8bf*)(Wt + (size_t)(bg * NNODE + col0 + bc) * K + kc + bp_);
  };

  stageA(0, 0);
  stageB(0, 0);
  __syncthreads();
  for (int kc = 0; kc < K; kc += 32) {
    const int cur = (kc >> 5) & 1, nxt = cur ^ 1;
    if (kc + 32 < K) {
      stageA(kc + 32, nxt);
      stageB(kc + 32, nxt);
    }
    const bf16_t* ap2 =
        sA[cur] + (wave * 16 + (lane & 15)) * 32 + ((lane >> 4) * 16);
    v16bf afrag = frag_cat(*(const v8bf*)ap2, *(const v8bf*)(ap2 + 8));
    v16bf bf[4];
#pragma unroll
    for (int g = 0; g < 4; ++g)
      bf[g] = *(const v16bf*)(sB[cur] + (g * 16 + (lane & 15)) * 32 +
                              ((lane >> 4) * 16));
#pragma unroll
    for (int g = 0; g < 4; ++g) acc[g] = wmma_bf16(afrag, bf[g], acc[g]);
    __syncthreads();
  }

  // Epilogue: accumulators for the 4 gates share the same (row, col) mapping.
  const int jg = col0 + (lane & 15);  // global hidden column
  const int rb = row0 + wave * 16 + ((lane >> 4) * 8);
  const float bf_ = bias[0 * NNODE + jg];
  const float bi_ = bias[1 * NNODE + jg];
  const float bo_ = bias[2 * NNODE + jg];
  const float bg_ = bias[3 * NNODE + jg];
#pragma unroll
  for (int v = 0; v < 8; ++v) {
    size_t idx = (size_t)(rb + v) * NNODE + jg;
    float f = fast_sigmoid(acc[0][v] + bf_);
    float i = fast_sigmoid(acc[1][v] + bi_);
    float o = fast_sigmoid(acc[2][v] + bo_);
    float g = fast_tanh(acc[3][v] + bg_);
    float cn = fmaf(f, c[idx], i * g);
    c[idx] = cn;
    hout[idx] = (bf16_t)(o * fast_tanh(cn));
  }
}

// ---------------------------------------------------------------------------
extern "C" void kernel_launch(void* const* d_in, const int* in_sizes, int n_in,
                              void* d_out, int out_size, void* d_ws,
                              size_t ws_size, hipStream_t stream) {
  (void)in_sizes; (void)n_in; (void)out_size; (void)ws_size;
  const float* inputs = (const float*)d_in[0];  // (T,B,V)
  const float* emb    = (const float*)d_in[1];  // (V,EMB)
  const float* W0     = (const float*)d_in[2];  // (EMB+NN, 4NN)
  const float* b0     = (const float*)d_in[3];
  const float* W1     = (const float*)d_in[4];  // (2NN, 4NN)
  const float* b1     = (const float*)d_in[5];
  const float* oW0    = (const float*)d_in[6];  // (NN, OUTN)
  const float* ob0    = (const float*)d_in[7];
  const float* oW1    = (const float*)d_in[8];  // (OUTN, V)
  const float* ob1    = (const float*)d_in[9];

  char* ws = (char*)d_ws;
  size_t off = 0;
  auto alloc = [&](size_t bytes) {
    void* p = ws + off;
    off += (bytes + 255) & ~(size_t)255;
    return p;
  };

  const size_t MB = (size_t)T_STEPS * BATCH;  // 51200 flattened rows
  bf16_t* Xinb = (bf16_t*)alloc(MB * VOCAB * 2);
  bf16_t* embt = (bf16_t*)alloc((size_t)EMBD * VOCAB * 2);
  bf16_t* W0t  = (bf16_t*)alloc((size_t)GATES * (EMBD + NNODE) * 2);
  bf16_t* W1t  = (bf16_t*)alloc((size_t)GATES * (2 * NNODE) * 2);
  bf16_t* oW0t = (bf16_t*)alloc((size_t)OUTN * NNODE * 2);
  bf16_t* oW1t = (bf16_t*)alloc((size_t)VOCAB * OUTN * 2);
  bf16_t* Xemb = (bf16_t*)alloc(MB * EMBD * 2);
  bf16_t* H1   = (bf16_t*)alloc(MB * NNODE * 2);
  bf16_t* H2   = (bf16_t*)alloc(MB * NNODE * 2);
  bf16_t* HS   = (bf16_t*)alloc(MB * OUTN * 2);
  float*  c0   = (float*)alloc((size_t)BATCH * NNODE * 4);
  float*  c1   = (float*)alloc((size_t)BATCH * NNODE * 4);
  bf16_t* hz   = (bf16_t*)alloc((size_t)BATCH * NNODE * 2);

  hipMemsetAsync(c0, 0, (size_t)BATCH * NNODE * 4, stream);
  hipMemsetAsync(c1, 0, (size_t)BATCH * NNODE * 4, stream);
  hipMemsetAsync(hz, 0, (size_t)BATCH * NNODE * 2, stream);

  {  // convert inputs to bf16
    int n = (int)(MB * VOCAB);
    k_cvt_bf16<<<(n + 255) / 256, 256, 0, stream>>>(inputs, Xinb, n);
  }
  {  // transpose+convert all weight matrices to (N,K) bf16
    long n;
    n = (long)VOCAB * EMBD;
    k_transpose_cvt<<<(unsigned)((n + 255) / 256), 256, 0, stream>>>(
        emb, embt, VOCAB, EMBD);
    n = (long)(EMBD + NNODE) * GATES;
    k_transpose_cvt<<<(unsigned)((n + 255) / 256), 256, 0, stream>>>(
        W0, W0t, EMBD + NNODE, GATES);
    n = (long)(2 * NNODE) * GATES;
    k_transpose_cvt<<<(unsigned)((n + 255) / 256), 256, 0, stream>>>(
        W1, W1t, 2 * NNODE, GATES);
    n = (long)NNODE * OUTN;
    k_transpose_cvt<<<(unsigned)((n + 255) / 256), 256, 0, stream>>>(
        oW0, oW0t, NNODE, OUTN);
    n = (long)OUTN * VOCAB;
    k_transpose_cvt<<<(unsigned)((n + 255) / 256), 256, 0, stream>>>(
        oW1, oW1t, OUTN, VOCAB);
  }

  // embeddings = inputs @ emb   (51200 x 256) = (51200 x 256)(256 x 256)
  k_gemm_wmma<0, 1><<<dim3((unsigned)(MB / 128), VOCAB / 64), 256, 0, stream>>>(
      Xinb, embt, nullptr, Xemb, (int)MB, EMBD, VOCAB);

  // LSTM layer 0 scan
  for (int t = 0; t < T_STEPS; ++t) {
    const bf16_t* hp = (t == 0) ? hz : (H1 + (size_t)(t - 1) * BATCH * NNODE);
    k_lstm_step<<<dim3(BATCH / 128, NNODE / 16), 256, 0, stream>>>(
        Xemb + (size_t)t * BATCH * EMBD, hp, W0t, b0, c0,
        H1 + (size_t)t * BATCH * NNODE, EMBD);
  }
  // LSTM layer 1 scan
  for (int t = 0; t < T_STEPS; ++t) {
    const bf16_t* hp = (t == 0) ? hz : (H2 + (size_t)(t - 1) * BATCH * NNODE);
    k_lstm_step<<<dim3(BATCH / 128, NNODE / 16), 256, 0, stream>>>(
        H1 + (size_t)t * BATCH * NNODE, hp, W1t, b1, c1,
        H2 + (size_t)t * BATCH * NNODE, NNODE);
  }

  // HS = relu(H2 @ oW0 + ob0)
  k_gemm_wmma<1, 1><<<dim3((unsigned)(MB / 128), OUTN / 64), 256, 0, stream>>>(
      H2, oW0t, ob0, HS, (int)MB, OUTN, NNODE);
  // logits = HS @ oW1 + ob1  (f32 output)
  k_gemm_wmma<0, 0><<<dim3((unsigned)(MB / 128), VOCAB / 64), 256, 0, stream>>>(
      HS, oW1t, ob1, d_out, (int)MB, VOCAB, OUTN);
}
